// DebugAttnProcessor_26929444946239
// MI455X (gfx1250) — compile-verified, matrix operand order checked by reading, not run on hardware
//
#include <hip/hip_runtime.h>
#include <hip/hip_bf16.h>

typedef __bf16 bf16_t;
typedef __attribute__((ext_vector_type(4)))  __bf16 v4bf;
typedef __attribute__((ext_vector_type(8)))  __bf16 v8bf;
typedef __attribute__((ext_vector_type(16))) __bf16 v16bf;
typedef __attribute__((ext_vector_type(8)))  float  v8f;
typedef __attribute__((ext_vector_type(4)))  float  v4f;

#define S_TOK 4096   // T_OUT * L
#define DMODEL 512
#define NH 8
#define HD 64

// ---------------------------------------------------------------------------
// WMMA helpers
// ---------------------------------------------------------------------------
__device__ __forceinline__ v8f wmma_bf16(v16bf a, v16bf b, v8f c) {
  // D = A(16x32) * B(32x16) + C, fp32 accumulate
  return __builtin_amdgcn_wmma_f32_16x16x32_bf16(
      /*neg_a=*/false, a, /*neg_b=*/false, b,
      /*c_mod=*/(short)0, c, /*reuse_a=*/false, /*reuse_b=*/false);
}

// Wave32 16-bit A and B fragments share the same per-lane pattern:
//   row (A:M / B:N) = lane&15
//   k elements: e<8 -> 8*(lane>>4)+e ; e>=8 -> 16 + 8*(lane>>4) + (e-8)
// Two 16-byte loads per lane.
__device__ __forceinline__ v16bf load_frag(const bf16_t* base, int stride) {
  const int lane = threadIdx.x & 31;
  const bf16_t* p = base + (size_t)(lane & 15) * stride + ((lane >> 4) << 3);
  v8bf lo = *(const v8bf*)p;
  v8bf hi = *(const v8bf*)(p + 16);
  return __builtin_shufflevector(lo, hi, 0,1,2,3,4,5,6,7,8,9,10,11,12,13,14,15);
}

// ---------------------------------------------------------------------------
// K0: convert x -> bf16, build transposed bf16 weights WT[w][n][k] = W[k][n]
// ---------------------------------------------------------------------------
__global__ __launch_bounds__(256) void convert_kernel(
    const float* __restrict__ x,
    const float* __restrict__ Wq, const float* __restrict__ Wk,
    const float* __restrict__ Wv, const float* __restrict__ Wo,
    bf16_t* __restrict__ xbf, bf16_t* __restrict__ WT) {
  const int i = blockIdx.x * 256 + threadIdx.x;
  const int XN = S_TOK * DMODEL;
  if (i < XN) {
    xbf[i] = (bf16_t)x[i];
    return;
  }
  const int j = i - XN;
  if (j >= 4 * DMODEL * DMODEL) return;
  const int w = j >> 18;                       // 512*512 = 262144 per matrix
  const int r = j & (DMODEL * DMODEL - 1);
  const int n = r >> 9;
  const int k = r & (DMODEL - 1);
  const float* W = (w == 0) ? Wq : (w == 1) ? Wk : (w == 2) ? Wv : Wo;
  WT[(size_t)w * DMODEL * DMODEL + (size_t)n * DMODEL + k] =
      (bf16_t)W[(size_t)k * DMODEL + n];
}

// ---------------------------------------------------------------------------
// K1: QKV projection.  qf/kf/vf[s][n] = sum_k xbf[s][k] * W[k][n]  (fp32 out)
// grid (S/256, D/64, 3), block 256 (8 waves); wave -> 32x64 tile.
// Fully unrolled k-loop: the compiler pipelines fragment loads across
// iterations itself (manual rotation regresses into v_mov storms).
// ---------------------------------------------------------------------------
__global__ __launch_bounds__(256) void gemm_qkv_kernel(
    const bf16_t* __restrict__ xbf, const bf16_t* __restrict__ WT,
    float* __restrict__ qf, float* __restrict__ kf, float* __restrict__ vf) {
  const int wave = threadIdx.x >> 5;
  const int lane = threadIdx.x & 31;
  const int m0 = (blockIdx.x * 8 + wave) * 32;
  const int n0 = blockIdx.y * 64;
  const int z = blockIdx.z;
  const bf16_t* BT = WT + (size_t)z * DMODEL * DMODEL;   // [n][k]
  float* out = (z == 0) ? qf : (z == 1) ? kf : vf;

  v8f zf = {0.f,0.f,0.f,0.f,0.f,0.f,0.f,0.f};
  v8f acc[2][4] = {{zf, zf, zf, zf}, {zf, zf, zf, zf}};

#pragma unroll
  for (int kk = 0; kk < DMODEL; kk += 32) {
    v16bf a0 = load_frag(xbf + (size_t)m0 * DMODEL + kk, DMODEL);
    v16bf a1 = load_frag(xbf + (size_t)(m0 + 16) * DMODEL + kk, DMODEL);
#pragma unroll
    for (int t = 0; t < 4; ++t) {
      v16bf b = load_frag(BT + (size_t)(n0 + 16 * t) * DMODEL + kk, DMODEL);
      acc[0][t] = wmma_bf16(a0, b, acc[0][t]);
      acc[1][t] = wmma_bf16(a1, b, acc[1][t]);
    }
  }
  const int col0 = n0 + (lane & 15);
#pragma unroll
  for (int mt = 0; mt < 2; ++mt) {
    const int rowb = m0 + mt * 16 + ((lane >> 4) << 3);
#pragma unroll
    for (int t = 0; t < 4; ++t)
#pragma unroll
      for (int r = 0; r < 8; ++r)
        out[(size_t)(rowb + r) * DMODEL + col0 + 16 * t] = acc[mt][t][r];
  }
}

// ---------------------------------------------------------------------------
// K2a: CAPE 4x4 transform + head-major bf16:  dst[h][s][hd]
//   out[s, g*4+j] = sc * sum_i src[s, g*4+i] * P[frame(s)][i][j]
// ---------------------------------------------------------------------------
__global__ __launch_bounds__(256) void cape_kernel(
    const float* __restrict__ src, const float* __restrict__ P, float sc,
    bf16_t* __restrict__ dst) {
  const int i = blockIdx.x * 256 + threadIdx.x;       // S * 128 threads
  if (i >= S_TOK * 128) return;
  const int s = i >> 7;
  const int g = i & 127;
  v4f a = *(const v4f*)(src + (size_t)s * DMODEL + g * 4);
  const float* p = P + ((s >> 9) << 4);               // frame = s / 512
  const int h  = g >> 4;
  const int hd = (g & 15) << 2;
  v4bf o;
#pragma unroll
  for (int j = 0; j < 4; ++j) {
    float v = a[0] * p[j] + a[1] * p[4 + j] + a[2] * p[8 + j] + a[3] * p[12 + j];
    o[j] = (bf16_t)(v * sc);
  }
  *(v4bf*)(dst + ((size_t)(h * S_TOK + s) * HD) + hd) = o;
}

// K2b: V transpose to vT[h][hd][S]  (== vT[col][s], col = h*64+hd)
__global__ __launch_bounds__(256) void vtrans_kernel(
    const float* __restrict__ vf, bf16_t* __restrict__ vT) {
  const int i = blockIdx.x * 256 + threadIdx.x;       // D * S threads
  if (i >= DMODEL * S_TOK) return;
  const int s = i & (S_TOK - 1);
  const int c = i >> 12;
  vT[(size_t)c * S_TOK + s] = (bf16_t)vf[(size_t)s * DMODEL + c];
}

// ---------------------------------------------------------------------------
// K3: flash attention (transposed).  Per wave: 32 queries x full hd=64.
//  St = K * Q^T      (keys = M rows, queries = N cols -> stats lane-local)
//  O^T = V^T * P^T   (P^T B-frag == St C-frag, register-for-register)
// Q was pre-scaled by hd^-0.5 * log2(e), so p = exp2(s - m).
// Two query tiles share every K/V fragment: 16 WMMA per 16 x b128 loads.
// grid (S/32/4, H), block 128
// ---------------------------------------------------------------------------
__global__ __launch_bounds__(128) void flash_kernel(
    const bf16_t* __restrict__ qh, const bf16_t* __restrict__ kh,
    const bf16_t* __restrict__ vT, bf16_t* __restrict__ obf) {
  const int wave = threadIdx.x >> 5;
  const int lane = threadIdx.x & 31;
  const int h  = blockIdx.y;
  const int q0 = (blockIdx.x * 4 + wave) * 32;

  const bf16_t* Q  = qh + ((size_t)h * S_TOK + q0) * HD;
  const bf16_t* Kh = kh + (size_t)h * S_TOK * HD;
  const bf16_t* Vh = vT + (size_t)h * HD * S_TOK;

  // Q^T B-fragments: [q-tile][hd chunk 0/1]
  v16bf Qb[2][2];
  Qb[0][0] = load_frag(Q, HD);
  Qb[0][1] = load_frag(Q + 32, HD);
  Qb[1][0] = load_frag(Q + 16 * HD, HD);
  Qb[1][1] = load_frag(Q + 16 * HD + 32, HD);

  v8f zf = {0.f,0.f,0.f,0.f,0.f,0.f,0.f,0.f};
  v8f Ot[2][4] = {{zf, zf, zf, zf}, {zf, zf, zf, zf}};
  float m[2]    = {-1e30f, -1e30f};
  float lsum[2] = {0.f, 0.f};

  for (int kb = 0; kb < S_TOK; kb += 32) {
    const bf16_t* Kp = Kh + (size_t)kb * HD;
    // K A-fragments: keys [kb,kb+16) x hd{0,1}, keys [kb+16,kb+32) x hd{0,1}
    v16bf Ka[4];
    Ka[0] = load_frag(Kp, HD);
    Ka[1] = load_frag(Kp + 32, HD);
    Ka[2] = load_frag(Kp + 16 * HD, HD);
    Ka[3] = load_frag(Kp + 16 * HD + 32, HD);
    // V^T A-fragments: hd tiles x 32 keys (shared by both q-tiles)
    v16bf Va[4];
#pragma unroll
    for (int t = 0; t < 4; ++t)
      Va[t] = load_frag(Vh + (size_t)(t * 16) * S_TOK + kb, S_TOK);

#pragma unroll
    for (int qt = 0; qt < 2; ++qt) {
      v8f st0 = zf, st1 = zf;
      st0 = wmma_bf16(Ka[0], Qb[qt][0], st0);
      st0 = wmma_bf16(Ka[1], Qb[qt][1], st0);
      st1 = wmma_bf16(Ka[2], Qb[qt][0], st1);
      st1 = wmma_bf16(Ka[3], Qb[qt][1], st1);

      // online softmax; per-query stats live in lane pairs (l, l^16)
      float mx = m[qt];
#pragma unroll
      for (int r = 0; r < 8; ++r) mx = fmaxf(mx, fmaxf(st0[r], st1[r]));
      mx = fmaxf(mx, __shfl_xor(mx, 16, 32));
      const float alpha = exp2f(m[qt] - mx);
      float ps = 0.f;
#pragma unroll
      for (int r = 0; r < 8; ++r) {
        st0[r] = exp2f(st0[r] - mx);
        st1[r] = exp2f(st1[r] - mx);
        ps += st0[r] + st1[r];
      }
      ps += __shfl_xor(ps, 16, 32);
      lsum[qt] = lsum[qt] * alpha + ps;
      m[qt] = mx;
#pragma unroll
      for (int t = 0; t < 4; ++t)
#pragma unroll
        for (int r = 0; r < 8; ++r) Ot[qt][t][r] *= alpha;

      // P^T B-fragment straight from St registers (no cross-lane traffic)
      v16bf Pb;
#pragma unroll
      for (int r = 0; r < 8; ++r) {
        Pb[r]     = (bf16_t)st0[r];
        Pb[8 + r] = (bf16_t)st1[r];
      }
#pragma unroll
      for (int t = 0; t < 4; ++t)
        Ot[qt][t] = wmma_bf16(Va[t], Pb, Ot[qt][t]);
    }
  }

#pragma unroll
  for (int qt = 0; qt < 2; ++qt) {
    const float inv = 1.0f / lsum[qt];
    const int token = q0 + qt * 16 + (lane & 15);
    const int colb  = h * HD + ((lane >> 4) << 3);
#pragma unroll
    for (int t = 0; t < 4; ++t) {
      v8bf pk;
#pragma unroll
      for (int r = 0; r < 8; ++r) pk[r] = (bf16_t)(Ot[qt][t][r] * inv);
      *(v8bf*)(obf + (size_t)token * DMODEL + colb + 16 * t) = pk;
    }
  }
}

// ---------------------------------------------------------------------------
// K4: out = o @ Wo + bo + residual   (fp32 output), wave -> 32x64 tile
// ---------------------------------------------------------------------------
__global__ __launch_bounds__(256) void out_gemm_kernel(
    const bf16_t* __restrict__ obf, const bf16_t* __restrict__ WoT,
    const float* __restrict__ bo, const float* __restrict__ resid,
    float* __restrict__ out) {
  const int wave = threadIdx.x >> 5;
  const int lane = threadIdx.x & 31;
  const int m0 = (blockIdx.x * 8 + wave) * 32;
  const int n0 = blockIdx.y * 64;

  v8f zf = {0.f,0.f,0.f,0.f,0.f,0.f,0.f,0.f};
  v8f acc[2][4] = {{zf, zf, zf, zf}, {zf, zf, zf, zf}};

#pragma unroll
  for (int kk = 0; kk < DMODEL; kk += 32) {
    v16bf a0 = load_frag(obf + (size_t)m0 * DMODEL + kk, DMODEL);
    v16bf a1 = load_frag(obf + (size_t)(m0 + 16) * DMODEL + kk, DMODEL);
#pragma unroll
    for (int t = 0; t < 4; ++t) {
      v16bf b = load_frag(WoT + (size_t)(n0 + 16 * t) * DMODEL + kk, DMODEL);
      acc[0][t] = wmma_bf16(a0, b, acc[0][t]);
      acc[1][t] = wmma_bf16(a1, b, acc[1][t]);
    }
  }
  const int col0 = n0 + (lane & 15);
#pragma unroll
  for (int mt = 0; mt < 2; ++mt) {
    const int rowb = m0 + mt * 16 + ((lane >> 4) << 3);
#pragma unroll
    for (int t = 0; t < 4; ++t) {
      const int col = col0 + 16 * t;
      const float bias = bo[col];
#pragma unroll
      for (int r = 0; r < 8; ++r) {
        const size_t idx = (size_t)(rowb + r) * DMODEL + col;
        out[idx] = acc[mt][t][r] + bias + resid[idx];
      }
    }
  }
}

// ---------------------------------------------------------------------------
// Host launcher
// ---------------------------------------------------------------------------
extern "C" void kernel_launch(void* const* d_in, const int* in_sizes, int n_in,
                              void* d_out, int out_size, void* d_ws, size_t ws_size,
                              hipStream_t stream) {
  const float* x         = (const float*)d_in[0];   // (8,512,512)
  const float* p_out     = (const float*)d_in[1];   // (1,8,4,4)
  const float* p_out_inv = (const float*)d_in[2];   // (1,8,4,4)
  const float* Wq        = (const float*)d_in[3];
  const float* Wk        = (const float*)d_in[4];
  const float* Wv        = (const float*)d_in[5];
  const float* Wo        = (const float*)d_in[6];
  const float* bo        = (const float*)d_in[7];
  float* out             = (float*)d_out;

  char* ws = (char*)d_ws;
  const size_t SD  = (size_t)S_TOK * DMODEL;   // 2M elems
  const size_t DD  = (size_t)DMODEL * DMODEL;  // 256K elems
  bf16_t* xbf = (bf16_t*)(ws);                       // 4 MB
  bf16_t* WT  = (bf16_t*)(ws + SD * 2);              // 4 * 0.5 MB (q,k,v,o)
  float*  qf  = (float*)(ws + SD * 2 + 4 * DD * 2);  // 8 MB
  float*  kf  = qf + SD;
  float*  vf  = kf + SD;
  bf16_t* qh  = (bf16_t*)(vf + SD);                  // [h][S][64] bf16, 4 MB
  bf16_t* kh  = qh + SD;
  bf16_t* vT  = kh + SD;                             // [h][hd][S]
  bf16_t* obf = vT + SD;                             // [s][d] bf16

  const float scale_q = 0.125f * 1.4426950408889634f;  // hd^-0.5 * log2(e)

  // K0: conversions / weight transposes
  {
    const int total = S_TOK * DMODEL + 4 * DMODEL * DMODEL;
    convert_kernel<<<(total + 255) / 256, 256, 0, stream>>>(
        x, Wq, Wk, Wv, Wo, xbf, WT);
  }
  // K1: QKV GEMMs
  {
    dim3 grid(S_TOK / 256, DMODEL / 64, 3);
    gemm_qkv_kernel<<<grid, 256, 0, stream>>>(xbf, WT, qf, kf, vf);
  }
  // K2: CAPE on q (scaled, p_out_inv) and k (p_out); V transpose
  {
    const int nt = S_TOK * 128;
    cape_kernel<<<(nt + 255) / 256, 256, 0, stream>>>(qf, p_out_inv, scale_q, qh);
    cape_kernel<<<(nt + 255) / 256, 256, 0, stream>>>(kf, p_out, 1.0f, kh);
    const int nv = DMODEL * S_TOK;
    vtrans_kernel<<<(nv + 255) / 256, 256, 0, stream>>>(vf, vT);
  }
  // K3: flash attention
  {
    dim3 grid(S_TOK / 32 / 4, NH);
    flash_kernel<<<grid, 128, 0, stream>>>(qh, kh, vT, obf);
  }
  // K4: output projection + bias + residual
  {
    dim3 grid(S_TOK / 256, DMODEL / 64);
    out_gemm_kernel<<<grid, 256, 0, stream>>>(
        obf, WT + 3 * DD, bo, x, out);
  }
}
